// nfpConv_18872086298717
// MI455X (gfx1250) — matrix operand order repaired and checked by reading, not compiled, not run on hardware
//
#include <hip/hip_runtime.h>

// ---------------------------------------------------------------------------
// Problem constants (match reference: B,A,D,F,O,NB = 64,1024,5,256,256,6)
// ---------------------------------------------------------------------------
#define BB   64
#define AA   1024
#define DD   5
#define FF   256
#define OO   256
#define NBB  6
#define KV   (FF + NBB)   // 262 real K
#define KP   288          // K padded to multiple of 32 (9 WMMA K-steps)
#define KPL  296          // LDS row stride (elements) — bank-conflict padding
#define WT_ELEMS (6 * OO * KP)

typedef __bf16 bf16_t;
typedef bf16_t v16bf __attribute__((ext_vector_type(16)));
typedef float  v8f   __attribute__((ext_vector_type(8)));

union V16 { v16bf v; uint4 q[2]; };

static __device__ __forceinline__ uint4 sel4(uint4 v, bool act) {
  uint4 r;
  r.x = act ? v.x : 0u; r.y = act ? v.y : 0u;
  r.z = act ? v.z : 0u; r.w = act ? v.w : 0u;
  return r;
}

// Build a 16-element bf16 fragment from two 16-byte chunks; zero if !act.
static __device__ __forceinline__ v16bf load_frag(const unsigned short* p0,
                                                  const unsigned short* p1,
                                                  bool act) {
  V16 t;
  t.q[0] = sel4(*(const uint4*)p0, act);
  t.q[1] = sel4(*(const uint4*)p1, act);
  return t.v;
}

static __device__ __forceinline__ void split_bf16(float x,
                                                  unsigned short& h,
                                                  unsigned short& l) {
  bf16_t hb = (bf16_t)x;             // RNE truncate to bf16 (hi 8 mantissa bits)
  float  hf = (float)hb;
  bf16_t lb = (bf16_t)(x - hf);      // residual (next ~8 bits)
  h = __builtin_bit_cast(unsigned short, hb);
  l = __builtin_bit_cast(unsigned short, lb);
}

// ---------------------------------------------------------------------------
// Kernel 0: pack degW [6][262][256] f32 -> transposed hi/lo bf16 [6][256][288]
// so B fragments are contiguous per-lane 16-bf16 runs along K.
// ---------------------------------------------------------------------------
__global__ void pack_w(const float* __restrict__ degW,
                       unsigned short* __restrict__ wh,
                       unsigned short* __restrict__ wl) {
  int idx = blockIdx.x * 256 + threadIdx.x;
  if (idx >= WT_ELEMS) return;
  int k = idx % KP;
  int r = idx / KP;
  int n = r % OO;
  int d = r / OO;
  float x = (k < KV) ? degW[((size_t)d * KV + k) * OO + n] : 0.0f;
  unsigned short h, l;
  split_bf16(x, h, l);
  wh[idx] = h;
  wl[idx] = l;
}

// ---------------------------------------------------------------------------
// Main kernel: one workgroup (128 thr = 4 waves) per 16-atom M-tile.
// ---------------------------------------------------------------------------
__global__ void __launch_bounds__(128)
nfp_conv_kernel(const float* __restrict__ atoms,
                const float* __restrict__ bonds,
                const int*   __restrict__ edges,
                const float* __restrict__ bias,
                const unsigned short* __restrict__ wh,
                const unsigned short* __restrict__ wl,
                float* __restrict__ out) {
  __shared__ unsigned short sh[16 * KPL];   // feats hi (bf16 bits)
  __shared__ unsigned short slo[16 * KPL];  // feats lo (bf16 bits)
  __shared__ int sedge[16 * DD];
  __shared__ int sdeg[16];

  const int tid = threadIdx.x;
  const int tile = blockIdx.x;            // 4096 tiles
  const int b  = tile >> 6;               // batch (A/16 = 64 tiles per batch)
  const int a0 = (tile & 63) << 4;        // first atom row of tile

  // ---- Phase 1a: edges, degrees, bond sums, K padding (16 threads) --------
  if (tid < 16) {
    const int r = tid;
    const size_t ga = (size_t)b * AA + (a0 + r);
    int deg = 0;
    #pragma unroll
    for (int d = 0; d < DD; ++d) {
      int e = edges[ga * DD + d];
      sedge[r * DD + d] = e;
      deg += (e != -1) ? 1 : 0;
    }
    sdeg[r] = deg;

    float bs[NBB];
    #pragma unroll
    for (int j = 0; j < NBB; ++j) bs[j] = 0.0f;
    #pragma unroll
    for (int d = 0; d < DD; ++d) {
      const float* bp = bonds + (ga * DD + d) * NBB;
      #pragma unroll
      for (int j = 0; j < NBB; ++j) bs[j] += bp[j];
    }
    #pragma unroll
    for (int j = 0; j < NBB; ++j) {
      unsigned short h, l;
      split_bf16(bs[j], h, l);
      sh[r * KPL + FF + j] = h;
      slo[r * KPL + FF + j] = l;
    }
    for (int j = KV; j < KP; ++j) {       // zero K padding
      sh[r * KPL + j] = 0;
      slo[r * KPL + j] = 0;
    }
  }
  __syncthreads();

  // ---- Phase 1b: gather + sum atom features (8 threads per row) -----------
  {
    const int r  = tid >> 3;              // row 0..15
    const int fg = tid & 7;               // feature group
    const size_t rowbase = (size_t)b * AA;
    const float* selfp = atoms + (rowbase + (a0 + r)) * FF;

    const float* np[DD];
    float        nv[DD];
    #pragma unroll
    for (int d = 0; d < DD; ++d) {
      int e = sedge[r * DD + d];
      int ec = (e >= 0) ? e : 0;          // clamp; contribution zeroed below
      np[d] = atoms + (rowbase + ec) * FF;
      nv[d] = (e >= 0) ? 1.0f : 0.0f;
    }

    #pragma unroll
    for (int c = 0; c < 8; ++c) {
      const int f = fg * 32 + c * 4;
      float4 s = *(const float4*)(selfp + f);
      #pragma unroll
      for (int d = 0; d < DD; ++d) {
        float4 t = *(const float4*)(np[d] + f);
        s.x += nv[d] * t.x; s.y += nv[d] * t.y;
        s.z += nv[d] * t.z; s.w += nv[d] * t.w;
      }
      float v[4] = {s.x, s.y, s.z, s.w};
      #pragma unroll
      for (int i = 0; i < 4; ++i) {
        unsigned short h, l;
        split_bf16(v[i], h, l);
        sh[r * KPL + f + i] = h;
        slo[r * KPL + f + i] = l;
      }
    }
  }
  __syncthreads();

  // ---- Phase 2: degree-masked WMMA GEMM, 64 output cols per wave ----------
  const int lane = tid & 31;
  const int w    = tid >> 5;
  const int hi   = lane >> 4;             // half-wave select
  const int Ml   = lane & 15;             // row (A) or column (B/C)

  int pres = 0;
  #pragma unroll
  for (int r = 0; r < 16; ++r) pres |= 1 << sdeg[r];
  pres = __builtin_amdgcn_readfirstlane(pres);   // scalar branch -> EXEC stays full
  const int mydeg = sdeg[Ml];

  v8f acc[4];
  #pragma unroll
  for (int n = 0; n < 4; ++n) acc[n] = (v8f){0,0,0,0,0,0,0,0};

  for (int d = 0; d < 6; ++d) {
    if (!((pres >> d) & 1)) continue;     // typically only d=5 executes
    const bool act = (mydeg == d);        // mask rows not in this bucket
    const unsigned short* whd = wh + (size_t)d * OO * KP;
    const unsigned short* wld = wl + (size_t)d * OO * KP;

    for (int k0 = 0; k0 < KP; k0 += 32) {
      // A fragment: M = Ml; K = k0 + 16*hi + (0..15), contiguous
      const unsigned short* pah = &sh[Ml * KPL + k0 + 16 * hi];
      const unsigned short* pal = &slo[Ml * KPL + k0 + 16 * hi];
      v16bf a_h = load_frag(pah, pah + 8, act);
      v16bf a_l = load_frag(pal, pal + 8, act);

      #pragma unroll
      for (int n = 0; n < 4; ++n) {
        const int ncol = w * 64 + n * 16 + Ml;
        // B fragment: N = ncol; K = k0 + 16*hi + (0..15), contiguous in Wt
        const unsigned short* pbh = whd + (size_t)ncol * KP + k0 + 16 * hi;
        const unsigned short* pbl = wld + (size_t)ncol * KP + k0 + 16 * hi;
        v16bf b_h = load_frag(pbh, pbh + 8, true);
        v16bf b_l = load_frag(pbl, pbl + 8, true);

        // hi/lo decomposition: AhBh + AhBl + AlBh ~ f32 product, f32 accum
        acc[n] = __builtin_amdgcn_wmma_f32_16x16x32_bf16(
            false, a_h, false, b_h, (short)0, acc[n], false, false);
        acc[n] = __builtin_amdgcn_wmma_f32_16x16x32_bf16(
            false, a_h, false, b_l, (short)0, acc[n], false, false);
        acc[n] = __builtin_amdgcn_wmma_f32_16x16x32_bf16(
            false, a_l, false, b_h, (short)0, acc[n], false, false);
      }
    }
  }

  // ---- Epilogue: bias + sigmoid, coalesced row stores ---------------------
  #pragma unroll
  for (int n = 0; n < 4; ++n) {
    const int ncol = w * 64 + n * 16 + Ml;
    const float bv = bias[ncol];
    float* op = out + ((size_t)b * AA + a0) * OO + ncol;
    #pragma unroll
    for (int i = 0; i < 8; ++i) {
      const int row = i + 8 * hi;         // C/D layout: VGPR i -> M = i + 8*hi
      float x = acc[n][i] + bv;
      op[(size_t)row * OO] = 1.0f / (1.0f + __expf(-x));
    }
  }
}

// ---------------------------------------------------------------------------
// Host launcher
// ---------------------------------------------------------------------------
extern "C" void kernel_launch(void* const* d_in, const int* in_sizes, int n_in,
                              void* d_out, int out_size, void* d_ws, size_t ws_size,
                              hipStream_t stream) {
  const float* atoms = (const float*)d_in[0];   // [64,1024,256]
  const float* bonds = (const float*)d_in[1];   // [64,1024,5,6]
  const int*   edges = (const int*)d_in[2];     // [64,1024,5]
  const float* degW  = (const float*)d_in[3];   // [6,262,256]
  const float* bias  = (const float*)d_in[4];   // [1,256]
  float* out = (float*)d_out;                   // [64,1024,256]

  unsigned short* wh = (unsigned short*)d_ws;            // 6*256*288 u16
  unsigned short* wl = wh + WT_ELEMS;                    // +884736 bytes

  pack_w<<<(WT_ELEMS + 255) / 256, 256, 0, stream>>>(degW, wh, wl);

  const int tiles = BB * (AA / 16);  // 4096
  nfp_conv_kernel<<<tiles, 128, 0, stream>>>(atoms, bonds, edges, bias, wh, wl, out);
}
